// Convolution_15333033247052
// MI455X (gfx1250) — matrix operand order repaired and checked by reading, not compiled, not run on hardware
//
#include <hip/hip_runtime.h>

typedef __attribute__((ext_vector_type(16))) _Float16 v16h;
typedef __attribute__((ext_vector_type(8)))  _Float16 v8h;
typedef __attribute__((ext_vector_type(8)))  float    v8f;

#define GRID3 192
#define DIMF  80
#define KPAD  96
#define NTAP  125

// ---------------- prep kernels ----------------

__global__ void k_init_vol(int* __restrict__ vol, int total) {
  int i = blockIdx.x * blockDim.x + threadIdx.x;
  int stride = gridDim.x * blockDim.x;
  for (; i < total; i += stride) vol[i] = -1;
}

__global__ void k_scatter(const int* __restrict__ coords, int* __restrict__ vol, int n) {
  int i = blockIdx.x * blockDim.x + threadIdx.x;
  if (i < n) {
    int x = coords[3*i], y = coords[3*i+1], z = coords[3*i+2];
    vol[(x * GRID3 + y) * GRID3 + z] = i;
  }
}

// pack feats -> f16, padded to 96 cols, plus one all-zero row at index n
__global__ void k_pack_feats(const float* __restrict__ feats, _Float16* __restrict__ f16,
                             int n, long long total) {
  long long i = (long long)blockIdx.x * blockDim.x + threadIdx.x;
  if (i >= total) return;
  int row = (int)(i / KPAD);
  int col = (int)(i % KPAD);
  float v = (row < n && col < DIMF) ? feats[(long long)row * DIMF + col] : 0.0f;
  f16[i] = (_Float16)v;
}

// build the 125 dense 80x80 kernels in f32; self-connection folded into center tap (l=62)
__global__ void k_build_K(const float* __restrict__ weight,
                          const float* __restrict__ w_sc0,
                          const float* __restrict__ w_sc1,
                          float* __restrict__ Kf) {
  __shared__ float w[2304];
  int l = blockIdx.x;                 // lattice index (permutation cancels in the sum)
  int li = l / 25, lj = (l / 5) % 5, lk = l % 5;
  float dx = (float)(li - 2), dy = (float)(lj - 2), dz = (float)(lk - 2);
  float d = sqrtf(dx*dx + dy*dy + dz*dz);

  float emb[8];
  #pragma unroll
  for (int c = 0; c < 8; ++c) {
    float t = 3.6f * d - (float)(c + 1);   // (d - centers[c]) / step
    float t2 = t * t;
    emb[c] = (t2 < 1.0f) ? 1.14136f * expf(2.0f - 2.0f / (1.0f - t2)) : 0.0f;
  }
  for (int j = threadIdx.x; j < 2304; j += blockDim.x) {
    float s = 0.0f;
    #pragma unroll
    for (int r = 0; r < 8; ++r) s += emb[r] * weight[r * 2304 + j];
    w[j] = s * (1.0f / 125.0f);
  }
  __syncthreads();

  float inv = (d > 0.0f) ? 1.0f / fmaxf(d, 1e-9f) : 0.0f;
  float vv[3];
  vv[0] = 1.7320508075688772f * dx * inv;
  vv[1] = 1.7320508075688772f * dy * inv;
  vv[2] = 1.7320508075688772f * dz * inv;
  const float A  = 0.14433756729740643f;  // 1/sqrt(48)
  const float A3 = 1.0f / 12.0f;          // 1/sqrt(144)
  const bool  ctr = (l == 62);            // offset (0,0,0): always hits self

  for (int e = threadIdx.x; e < 6400; e += blockDim.x) {
    int a = e / 80, b = e % 80;
    float val;
    if (a < 32) {
      if (b < 32) {
        val = A * w[32 * a + b];                                     // K00
        if (ctr) val += 0.17677669529663687f * w_sc0[32 * a + b];    // sc0: w_sc0/sqrt(32)
      } else {
        int q = b - 32, kk = q / 3, m = q % 3;
        val = A * w[1024 + 16 * a + kk] * vv[m];                     // K01
      }
    } else {
      int p = a - 32, ii = p / 3, m = p % 3;
      if (b < 32) {
        val = A3 * w[1792 + 32 * ii + b] * vv[m];                    // K10
      } else {
        int q = b - 32, kk = q / 3, nn = q % 3;
        val = (m == nn) ? A * w[1536 + 16 * ii + kk] : 0.0f;         // K11
        if (ctr && m == nn) val += 0.25f * w_sc1[16 * ii + kk];      // sc1: w_sc1/sqrt(16)
      }
    }
    Kf[l * 6400 + e] = val;
  }
}

// Pre-swizzle K into WMMA B-fragment layout (16x16x32 f16):
// fragment id = l*15 + kc*5 + nt, 512 halves/fragment, lane owns 16 contiguous halves.
// element at (lane, h): K_global = kc*32 + (h>>3)*16 + (lane>>4)*8 + (h&7), N_global = nt*16 + (lane&15)
__global__ void k_pack_K(const float* __restrict__ Kf, _Float16* __restrict__ Kb) {
  int idx = blockIdx.x * blockDim.x + threadIdx.x;
  if (idx >= NTAP * 15 * 512) return;
  int frag = idx >> 9, slot = idx & 511;
  int lane = slot >> 4, h = slot & 15;
  int l = frag / 15, rem = frag % 15, kc = rem / 5, nt = rem % 5;
  int kg = kc * 32 + ((h >> 3) << 4) + ((lane >> 4) << 3) + (h & 7);
  int ng = nt * 16 + (lane & 15);
  float v = (kg < DIMF) ? Kf[l * 6400 + kg * 80 + ng] : 0.0f;
  Kb[idx] = (_Float16)v;
}

// ---------------- hot kernel: 125-tap WMMA convolution ----------------
// one wave32 per 32-row tile (two 16-row WMMA tiles sharing all B fragments)
__device__ __forceinline__ int nb_lookup(const int* __restrict__ vol, int l,
                                         int vb, int okx, int oky, int okz) {
  int i = l / 25, j = (l / 5) % 5, k = l % 5;
  int ok = (okx >> i) & (oky >> j) & (okz >> k) & 1;
  int voff = (i - 2) * (GRID3 * GRID3) + (j - 2) * GRID3 + (k - 2);
  int t = vol[vb + (ok ? voff : 0)];
  return ok ? t : -1;
}

__global__ void __launch_bounds__(256)
k_conv(const _Float16* __restrict__ f16,
       const _Float16* __restrict__ Kb,
       const int* __restrict__ coords,
       const int* __restrict__ vol,
       float* __restrict__ out, int n, int ntiles) {
  int wave = blockIdx.x * (blockDim.x >> 5) + (threadIdx.x >> 5);
  if (wave >= ntiles) return;                 // uniform per wave -> EXEC stays all-ones
  int lane = threadIdx.x & 31;
  int m    = lane & 15;
  int half = lane >> 4;
  int rowbase = wave << 5;                    // 32 rows per wave

  // per-lane rows for the two 16-row M-tiles
  int r0 = rowbase + m;
  int r1 = rowbase + 16 + m;
  int rc0 = (r0 < n) ? r0 : (n - 1);
  int rc1 = (r1 < n) ? r1 : (n - 1);
  int a0x = coords[3*rc0], a0y = coords[3*rc0+1], a0z = coords[3*rc0+2];
  int a1x = coords[3*rc1], a1y = coords[3*rc1+1], a1z = coords[3*rc1+2];

  // per-lane validity bitmasks (bit i <=> offset i-2 in range) + base volume index
  int okx0=0, oky0=0, okz0=0, okx1=0, oky1=0, okz1=0;
  #pragma unroll
  for (int t = 0; t < 5; ++t) {
    okx0 |= (((unsigned)(a0x + t - 2) < (unsigned)GRID3) ? 1 : 0) << t;
    oky0 |= (((unsigned)(a0y + t - 2) < (unsigned)GRID3) ? 1 : 0) << t;
    okz0 |= (((unsigned)(a0z + t - 2) < (unsigned)GRID3) ? 1 : 0) << t;
    okx1 |= (((unsigned)(a1x + t - 2) < (unsigned)GRID3) ? 1 : 0) << t;
    oky1 |= (((unsigned)(a1y + t - 2) < (unsigned)GRID3) ? 1 : 0) << t;
    okz1 |= (((unsigned)(a1z + t - 2) < (unsigned)GRID3) ? 1 : 0) << t;
  }
  int vb0 = (a0x * GRID3 + a0y) * GRID3 + a0z;
  int vb1 = (a1x * GRID3 + a1y) * GRID3 + a1z;

  v8f accA[5], accB[5];
  #pragma unroll
  for (int nt = 0; nt < 5; ++nt) { accA[nt] = (v8f){}; accB[nt] = (v8f){}; }

  // software pipeline: prefetch neighbor indices one tap ahead of the WMMA work
  int nb0 = nb_lookup(vol, 0, vb0, okx0, oky0, okz0);
  int nb1 = nb_lookup(vol, 0, vb1, okx1, oky1, okz1);

  for (int l = 0; l < NTAP; ++l) {
    int cur0 = nb0, cur1 = nb1;
    if (l + 1 < NTAP) {
      nb0 = nb_lookup(vol, l + 1, vb0, okx0, oky0, okz0);
      nb1 = nb_lookup(vol, l + 1, vb1, okx1, oky1, okz1);
    }
    long long arow0 = (cur0 >= 0) ? (long long)cur0 : (long long)n;  // row n = zero row
    long long arow1 = (cur1 >= 0) ? (long long)cur1 : (long long)n;
    const _Float16* ap0 = f16 + arow0 * KPAD + (half << 3);
    const _Float16* ap1 = f16 + arow1 * KPAD + (half << 3);
    const _Float16* bp  = Kb + (long long)l * (15 * 512) + (lane << 4);

    #pragma unroll
    for (int kc = 0; kc < 3; ++kc) {
      union { v16h v; v8h h2[2]; } a0, a1;
      a0.h2[0] = *(const v8h*)(ap0 + kc * 32);        // K = 8*half .. +7
      a0.h2[1] = *(const v8h*)(ap0 + kc * 32 + 16);   // K = 16+8*half .. +7
      a1.h2[0] = *(const v8h*)(ap1 + kc * 32);
      a1.h2[1] = *(const v8h*)(ap1 + kc * 32 + 16);
      const _Float16* bk = bp + kc * 5 * 512;
      #pragma unroll
      for (int nt = 0; nt < 5; ++nt) {
        v16h b = *(const v16h*)(bk + nt * 512);
        accA[nt] = __builtin_amdgcn_wmma_f32_16x16x32_f16(false, a0.v, false, b, (short)0, accA[nt], false, false);
        accB[nt] = __builtin_amdgcn_wmma_f32_16x16x32_f16(false, a1.v, false, b, (short)0, accB[nt], false, false);
      }
    }
  }

  // epilogue: pure store (sc folded into K[62]); VGPR rr -> M = 8*half + rr, N = lane&15
  #pragma unroll
  for (int nt = 0; nt < 5; ++nt) {
    #pragma unroll
    for (int rr = 0; rr < 8; ++rr) {
      int orow0 = rowbase + (half << 3) + rr;
      if (orow0 < n)
        out[(long long)orow0 * DIMF + nt * 16 + m] = accA[nt][rr];
      int orow1 = rowbase + 16 + (half << 3) + rr;
      if (orow1 < n)
        out[(long long)orow1 * DIMF + nt * 16 + m] = accB[nt][rr];
    }
  }
}

// ---------------- launcher ----------------

extern "C" void kernel_launch(void* const* d_in, const int* in_sizes, int n_in,
                              void* d_out, int out_size, void* d_ws, size_t ws_size,
                              hipStream_t stream) {
  const float* feats  = (const float*)d_in[0];
  const float* weight = (const float*)d_in[1];
  const float* w_sc0  = (const float*)d_in[2];
  const float* w_sc1  = (const float*)d_in[3];
  const int*   coords = (const int*)d_in[4];
  int n = in_sizes[0] / DIMF;
  float* out = (float*)d_out;

  char* ws = (char*)d_ws;
  size_t o = 0;
  int* vol = (int*)(ws + o);        o += (size_t)GRID3 * GRID3 * GRID3 * 4;
  o = (o + 255) & ~(size_t)255;
  _Float16* f16 = (_Float16*)(ws + o); o += (size_t)(n + 16) * KPAD * 2;
  o = (o + 255) & ~(size_t)255;
  float* Kf = (float*)(ws + o);     o += (size_t)NTAP * 6400 * 4;
  o = (o + 255) & ~(size_t)255;
  _Float16* Kb = (_Float16*)(ws + o); o += (size_t)NTAP * 15 * 512 * 2;
  (void)ws_size;

  k_init_vol<<<2048, 256, 0, stream>>>(vol, GRID3 * GRID3 * GRID3);
  k_scatter<<<(n + 255) / 256, 256, 0, stream>>>(coords, vol, n);
  long long fp = (long long)(n + 1) * KPAD;
  k_pack_feats<<<(int)((fp + 255) / 256), 256, 0, stream>>>(feats, f16, n, fp);
  k_build_K<<<NTAP, 256, 0, stream>>>(weight, w_sc0, w_sc1, Kf);
  k_pack_K<<<(NTAP * 15 * 512 + 255) / 256, 256, 0, stream>>>(Kf, Kb);

  int ntiles = (n + 31) / 32;                  // 32 rows per wave now
  int wavesPerBlock = 8;                       // 256 threads = 8 wave32
  int blocks = (ntiles + wavesPerBlock - 1) / wavesPerBlock;
  k_conv<<<blocks, 256, 0, stream>>>(f16, Kb, coords, vol, out, n, ntiles);
}
